// KernelDWConv2d_44727789420779
// MI455X (gfx1250) — matrix operand order repaired and checked by reading, not compiled, not run on hardware
//
#include <hip/hip_runtime.h>

// Modulated conv as per-batch implicit GEMM on bf16 WMMA:
//   out[b] (O x 625) = (weight .* kernel[b]) (O x 12544)  @  im2col(x[b]) (12544 x 625)
// Compute-bound (128 GFLOP, ~66MB unique data, L2-resident) -> bf16 WMMA, f32 acc.
//  - A fragments (modulated weight) built directly from global: each element is
//    consumed by exactly one wave, so LDS staging for A is pure overhead.
//  - B tile (im2col) double-buffered in LDS; next chunk's gathers are issued
//    before the current chunk's WMMAs (pipelined), packed+stored after them.
//  - im2col addressing: SALU-walked uniform base + loop-invariant VGPR pixel
//    offset (GVS global_load) -> no per-element VALU address math.
//  - WMMA order: k-step outer / n-tile inner, so the two WMMAs sharing an
//    accumulator are 8 independent WMMAs apart (no XDL dependency bubbles).

#define BATCH 32
#define CIN   256
#define COUT  256
#define HIN   31
#define HO    25
#define NPIX  625              // 25*25
#define KDIM  12544            // CIN * 7 * 7 = 196 * 64
#define KC    64               // K per iteration (2 x 32-wide WMMA)
#define NITER 196              // KDIM / KC
#define O_BLK 128
#define P_BLK 128
#define LDSW  72               // 64 + 8 halfword pad (144B rows, 16B aligned)

typedef __attribute__((ext_vector_type(16))) __bf16 v16bf;
typedef __attribute__((ext_vector_type(2)))  __bf16 v2bf;
typedef __attribute__((ext_vector_type(8)))  float  v8f;

__device__ __forceinline__ unsigned pack2(float a, float b) {
#if __has_builtin(__builtin_amdgcn_cvt_pk_bf16_f32)
  union { v2bf v; unsigned u; } r;
  r.v = __builtin_amdgcn_cvt_pk_bf16_f32(a, b);   // v_cvt_pk_bf16_f32
  return r.u;
#else
  unsigned ua = __float_as_uint(a);
  unsigned ub = __float_as_uint(b);
  ua += 0x7FFFu + ((ua >> 16) & 1u);              // RNE
  ub += 0x7FFFu + ((ub >> 16) & 1u);
  return (ua >> 16) | (ub & 0xFFFF0000u);
#endif
}

union FragU { v16bf v; uint4 u[2]; };

__device__ __forceinline__ v8f wmma_bf16(const FragU& a, const FragU& b, v8f c) {
  return __builtin_amdgcn_wmma_f32_16x16x32_bf16(
      false, a.v, false, b.v, (short)0, c, false, false);
}

__global__ __launch_bounds__(256)
void dwmodconv_wmma_bf16(const float* __restrict__ x,
                         const float* __restrict__ kern,
                         const float* __restrict__ weight,
                         const float* __restrict__ bias,
                         float* __restrict__ out) {
  // Double-buffered im2col tile (only B needs LDS: shared by all 8 waves).
  __shared__ __align__(16) unsigned short Xs[2][P_BLK * LDSW];

  const int tid  = threadIdx.x;
  const int lane = tid & 31;
  const int wv   = tid >> 5;            // 8 waves (wave32)
  const int hi   = (lane >> 4) & 1;     // lane >= 16
  const int ln   = lane & 15;

  const int pBase = blockIdx.x * P_BLK; // pixel tile (5 tiles, padded to 640)
  const int oBase = blockIdx.y * O_BLK; // out-channel tile (2 tiles)
  const int b     = blockIdx.z;

  // X loader: row = tid&127; K-half is WAVE-UNIFORM so the walk lives in SGPRs.
  const int lrow = tid & 127;           // 0..127
  const int lkk  = (tid >> 7) << 5;     // 0 or 32, uniform per wave

  const int  m      = pBase + lrow;
  const int  mclamp = (m < NPIX) ? m : (NPIX - 1);   // padded rows read row 624
  const int  oh     = mclamp / HO;
  const int  ow     = mclamp - oh * HO;
  const int  pixoff = oh * HIN + ow;    // divergent, loop-invariant (VGPR)
  const float* xb   = x + (size_t)b * CIN * (HIN * HIN);   // uniform (SGPR)

  // A side: this wave's weight row and the (lane-phased) kernel row.
  const int  ak   = hi ? 8 : 0;         // A frag: lanes<16 K{0..7,16..23}, >=16 {8..15,24..31}
  const float* wrow = weight + (size_t)(oBase + wv * 16 + ln) * KDIM + ak;
  const float* krow = kern   + (size_t)b * KDIM + ak;

  v8f acc[8] = {};

  // ---- im2col gather of one KC chunk into registers (SALU base walk) ----
  auto load_x = [&](int kc, float* xr) {
    int t0 = __builtin_amdgcn_readfirstlane(kc + lkk);
    int c  = t0 / 49;
    int rs = t0 - c * 49;
    int r  = rs / 7;
    int s  = rs - r * 7;
    const float* bp = xb + (c * (HIN * HIN) + r * HIN + s);  // uniform base
#pragma unroll
    for (int e = 0; e < 32; ++e) {
      xr[e] = bp[pixoff];                        // SADDR base + VGPR offset
      if (++s == 7) {
        s = 0; bp += (HIN - 6);                  // +25
        if (++r == 7) { r = 0; bp += (HIN * HIN - 7 * HIN); }  // +744
      } else {
        bp += 1;
      }
    }
  };
  auto store_x = [&](const float* xr, int buf) {
#pragma unroll
    for (int q = 0; q < 8; ++q) {
      uint2 pr;
      pr.x = pack2(xr[4 * q],     xr[4 * q + 1]);
      pr.y = pack2(xr[4 * q + 2], xr[4 * q + 3]);
      *(uint2*)&Xs[buf][lrow * LDSW + lkk + 4 * q] = pr;
    }
  };
  // ---- A-frag build (direct from global) + 16 WMMAs from Xs[buf] ----
  auto compute_chunk = [&](int kc, int buf) {
    const float4* wp = (const float4*)(wrow + kc);
    const float4* kp = (const float4*)(krow + kc);
    FragU a0, a1;
    {
      float4 w0 = wp[0],  w1 = wp[1],  w2 = wp[4],  w3 = wp[5];
      float4 g0 = kp[0],  g1 = kp[1],  g2 = kp[4],  g3 = kp[5];
      a0.u[0].x = pack2(w0.x * g0.x, w0.y * g0.y);
      a0.u[0].y = pack2(w0.z * g0.z, w0.w * g0.w);
      a0.u[0].z = pack2(w1.x * g1.x, w1.y * g1.y);
      a0.u[0].w = pack2(w1.z * g1.z, w1.w * g1.w);
      a0.u[1].x = pack2(w2.x * g2.x, w2.y * g2.y);
      a0.u[1].y = pack2(w2.z * g2.z, w2.w * g2.w);
      a0.u[1].z = pack2(w3.x * g3.x, w3.y * g3.y);
      a0.u[1].w = pack2(w3.z * g3.z, w3.w * g3.w);
    }
    {
      float4 w0 = wp[8],  w1 = wp[9],  w2 = wp[12], w3 = wp[13];
      float4 g0 = kp[8],  g1 = kp[9],  g2 = kp[12], g3 = kp[13];
      a1.u[0].x = pack2(w0.x * g0.x, w0.y * g0.y);
      a1.u[0].y = pack2(w0.z * g0.z, w0.w * g0.w);
      a1.u[0].z = pack2(w1.x * g1.x, w1.y * g1.y);
      a1.u[0].w = pack2(w1.z * g1.z, w1.w * g1.w);
      a1.u[1].x = pack2(w2.x * g2.x, w2.y * g2.y);
      a1.u[1].y = pack2(w2.z * g2.z, w2.w * g2.w);
      a1.u[1].z = pack2(w3.x * g3.x, w3.y * g3.y);
      a1.u[1].w = pack2(w3.z * g3.z, w3.w * g3.w);
    }
    if (kc + KC < KDIM) {
      __builtin_prefetch(wrow + kc + KC, 0, 1);   // global_prefetch_b8
      __builtin_prefetch(krow + kc + KC, 0, 1);
    }
    const int bk = hi ? 8 : 0;           // B frag: lanes<16 K{0..7,16..23}, >=16 {8..15,24..31}
#pragma unroll
    for (int ks = 0; ks < 2; ++ks) {     // k-step outer: dependent WMMAs 8 apart
      const FragU& a = ks ? a1 : a0;
#pragma unroll
      for (int nt = 0; nt < 8; ++nt) {
        const unsigned short* brow =
            &Xs[buf][(nt * 16 + ln) * LDSW + ks * 32 + bk];
        FragU bfr;
        bfr.u[0] = ((const uint4*)brow)[0];
        bfr.u[1] = ((const uint4*)brow)[2];
        acc[nt] = wmma_bf16(a, bfr, acc[nt]);
      }
    }
  };

  // ---- pipelined main loop: gather(i+1) || wmma(i), pack+store after wmma ----
  {
    float xr[32];
    load_x(0, xr);
    store_x(xr, 0);
  }
  __syncthreads();
  int cur = 0;
  for (int it = 0; it < NITER - 1; ++it) {
    float xr[32];
    load_x((it + 1) * KC, xr);       // issue gathers; completes under WMMAs
    compute_chunk(it * KC, cur);     // 16 WMMAs from current buffer
    store_x(xr, cur ^ 1);            // waits loads; fills other buffer
    __syncthreads();
    cur ^= 1;
  }
  compute_chunk((NITER - 1) * KC, cur);   // peeled last chunk

  // ---- epilogue: D layout -> lanes = consecutive pixels (coalesced stores) ----
  float* ob = out + (size_t)b * COUT * NPIX;
#pragma unroll
  for (int nt = 0; nt < 8; ++nt) {
    const int mm = pBase + nt * 16 + ln;
    if (mm < NPIX) {
#pragma unroll
      for (int j = 0; j < 8; ++j) {
        const int o = oBase + wv * 16 + j + (hi ? 8 : 0);
        ob[(size_t)o * NPIX + mm] = acc[nt][j] + bias[o];
      }
    }
  }
}

extern "C" void kernel_launch(void* const* d_in, const int* in_sizes, int n_in,
                              void* d_out, int out_size, void* d_ws, size_t ws_size,
                              hipStream_t stream) {
  const float* x      = (const float*)d_in[0];
  const float* kern   = (const float*)d_in[1];
  const float* weight = (const float*)d_in[2];
  const float* bias   = (const float*)d_in[3];
  float* out          = (float*)d_out;
  (void)in_sizes; (void)n_in; (void)out_size; (void)d_ws; (void)ws_size;

  dim3 grid((NPIX + P_BLK - 1) / P_BLK,   // 5 pixel tiles
            COUT / O_BLK,                 // 2 channel tiles
            BATCH);                       // 32 batches
  dwmodconv_wmma_bf16<<<grid, dim3(256), 0, stream>>>(x, kern, weight, bias, out);
}
